// GCN_LSTM_27968827032304
// MI455X (gfx1250) — matrix operand order repaired and checked by reading, not compiled, not run on hardware
//
#include <hip/hip_runtime.h>
#include <hip/hip_bf16.h>
#include <cstdint>

typedef __attribute__((ext_vector_type(16))) __bf16 v16bf;
typedef __attribute__((ext_vector_type(8)))  __bf16 v8bf;
typedef __attribute__((ext_vector_type(8)))  float  v8f;

__device__ __forceinline__ __bf16 f2bf(float f) {
  unsigned u = __builtin_bit_cast(unsigned, f);
  u += 0x7FFFu + ((u >> 16) & 1u);            // round-to-nearest-even
  unsigned short h = (unsigned short)(u >> 16);
  return __builtin_bit_cast(__bf16, h);
}
__device__ __forceinline__ float sigmf(float x) { return 1.0f / (1.0f + __expf(-x)); }

// ---------------------------------------------------------------------------
// WMMA GEMM: C[M,N] = A[M,K](f32,row-major) * Bt[Np,K]^T(bf16) (+bias)(+relu)
// Requirements (guaranteed by caller): M % 16 == 0, K % 32 == 0 (A padded),
// Bt is [Np,K] with Np % (16*NSUB) == 0, zero-padded rows for n >= N.
// One wave computes a 16(M) x (16*NSUB)(N) tile, K-loop in steps of 32.
// All WMMAs sit in uniform, EXEC-full control flow.
// ---------------------------------------------------------------------------
template<int NSUB>
__global__ void __launch_bounds__(128)
wmma_gemm(const float* __restrict__ A, const __bf16* __restrict__ Bt,
          const float* __restrict__ bias, float* __restrict__ C,
          int M, int K, int N, int Np, int relu)
{
  const int lane = threadIdx.x & 31;
  const int wave = __builtin_amdgcn_readfirstlane((int)(threadIdx.x >> 5)); // SGPR
  const int nUnitsM = M >> 4;
  const int nUnitsN = Np / (16 * NSUB);
  const int unit = blockIdx.x * 4 + wave;
  if (unit >= nUnitsM * nUnitsN) return;          // scalar branch, EXEC untouched
  const int mt = unit % nUnitsM;
  const int ng = unit / nUnitsM;
  const int mbase = mt << 4;
  const float* arow = A + (size_t)(mbase + (lane & 15)) * K;
  const int khalf = (lane >> 4) << 4;             // 0 or 16

  v8f acc[NSUB];
#pragma unroll
  for (int sIdx = 0; sIdx < NSUB; ++sIdx) acc[sIdx] = v8f{};

  const int nkt = K >> 5;
  for (int kt = 0; kt < nkt; ++kt) {
    const int k0 = kt << 5;
    const float4* p = (const float4*)(arow + k0 + khalf);
    const float4* q = (const float4*)(arow + k0 + 8 + khalf);
    float4 f0 = p[0], f1 = p[1], f2 = q[0], f3 = q[1];
    v16bf a;
    a[0]  = f2bf(f0.x); a[1]  = f2bf(f0.y); a[2]  = f2bf(f0.z); a[3]  = f2bf(f0.w);
    a[4]  = f2bf(f1.x); a[5]  = f2bf(f1.y); a[6]  = f2bf(f1.z); a[7]  = f2bf(f1.w);
    a[8]  = f2bf(f2.x); a[9]  = f2bf(f2.y); a[10] = f2bf(f2.z); a[11] = f2bf(f2.w);
    a[12] = f2bf(f3.x); a[13] = f2bf(f3.y); a[14] = f2bf(f3.z); a[15] = f2bf(f3.w);

    const __bf16* bb = Bt + (size_t)((ng * NSUB * 16) + (lane & 15)) * K + k0 + ((lane >> 4) << 4);
#pragma unroll
    for (int sIdx = 0; sIdx < NSUB; ++sIdx) {
      v16bf b = *(const v16bf*)(bb + (size_t)sIdx * 16 * K);
      acc[sIdx] = __builtin_amdgcn_wmma_f32_16x16x32_bf16(
          false, a, false, b, (short)0, acc[sIdx], false, false);
    }
  }
  const int mrow = mbase + ((lane >> 4) << 3);
#pragma unroll
  for (int sIdx = 0; sIdx < NSUB; ++sIdx) {
    int n = (ng * NSUB * 16) + (sIdx << 4) + (lane & 15);
    if (n < N) {
      float bv = bias ? bias[n] : 0.0f;
#pragma unroll
      for (int r = 0; r < 8; ++r) {
        float v = acc[sIdx][r] + bv;
        if (relu) v = fmaxf(v, 0.0f);
        C[(size_t)(mrow + r) * N + n] = v;
      }
    }
  }
}

// ---------------------------------------------------------------------------
// Weight prep: Bt[n][k] bf16, zero-padded.  direct=1: W is [N,K]; else W is [K,N].
// ---------------------------------------------------------------------------
__global__ void prep_weight(const float* __restrict__ W, __bf16* __restrict__ Bt,
                            int K, int N, int Kp, int Np, int direct)
{
  int idx = blockIdx.x * blockDim.x + threadIdx.x;
  if (idx >= Kp * Np) return;
  int k = idx % Kp, n = idx / Kp;
  float v = 0.0f;
  if (k < K && n < N) v = direct ? W[(size_t)n * K + k] : W[(size_t)k * N + n];
  Bt[idx] = f2bf(v);
}

// Pad x features [rows,5] -> [rows,32] fp32 (zero-filled) so the GEMM needs no K guards.
__global__ void pad_x5(const float* __restrict__ x, float* __restrict__ Xp, int rows) {
  int idx = blockIdx.x * blockDim.x + threadIdx.x;
  if (idx >= rows * 32) return;
  int r = idx >> 5, k = idx & 31;
  Xp[idx] = (k < 5) ? x[(size_t)r * 5 + k] : 0.0f;
}

__global__ void add2(const float* a, const float* b, float* o, int n) {
  int i = blockIdx.x * blockDim.x + threadIdx.x;
  if (i < n) o[i] = a[i] + b[i];
}
__global__ void zero_f(float* p, int n) {
  int i = blockIdx.x * blockDim.x + threadIdx.x;
  if (i < n) p[i] = 0.0f;
}
__global__ void count_deg(float* deg, const int* dst, int E, int NNc) {
  int e = blockIdx.x * blockDim.x + threadIdx.x;
  if (e >= E + NNc) return;
  int d = (e < E) ? dst[e] : (e - E);
  atomicAdd(&deg[d], 1.0f);
}
__global__ void edge_norm(float* nrm, const float* deg, const int* src, const int* dst,
                          int E, int NNc) {
  int e = blockIdx.x * blockDim.x + threadIdx.x;
  if (e >= E + NNc) return;
  int s = (e < E) ? src[e] : (e - E);
  int d = (e < E) ? dst[e] : (e - E);
  nrm[e] = rsqrtf(deg[s]) * rsqrtf(deg[d]);
}
__global__ void fill_bias(float* agg, const float* bias, int rows, int F) {
  int i = blockIdx.x * blockDim.x + threadIdx.x;
  if (i < rows * F) agg[i] = bias[i % F];
}
__global__ void scatter_edges(const float* __restrict__ Ms, float* __restrict__ agg,
                              const int* __restrict__ src, const int* __restrict__ dst,
                              const float* __restrict__ nrm, int nE, int E, int F)
{
  int idx = blockIdx.x * blockDim.x + threadIdx.x;
  if (idx >= nE * F) return;
  int e = idx / F, f = idx % F;
  int s = (e < E) ? src[e] : (e - E);
  int d = (e < E) ? dst[e] : (e - E);
  atomicAdd(&agg[(size_t)d * F + f], Ms[(size_t)s * F + f] * nrm[e]);
}
__global__ void relu_ip(float* a, int n) {
  int i = blockIdx.x * blockDim.x + threadIdx.x;
  if (i < n) a[i] = fmaxf(a[i], 0.0f);
}
__global__ void pool_frame(const float* __restrict__ A3, float* __restrict__ pooled_t,
                           int Bg, int Nj, int F) {
  int idx = blockIdx.x * blockDim.x + threadIdx.x;
  if (idx >= Bg * F) return;
  int b = idx / F, f = idx % F;
  float s = 0.0f;
  for (int n = 0; n < Nj; ++n) s += A3[(size_t)(b * Nj + n) * F + f];
  pooled_t[idx] = s / (float)Nj;
}

// ---------------------------------------------------------------------------
// Recurrent LSTM layer: single workgroup (8 waves) loops over S=256 steps.
// Gpre[(m*256+s)*4H + n] = x@Wih^T + bih + bhh (precomputed).
// h kept as bf16 in LDS (WMMA A-operand source), c as fp32 in LDS.
// Per step: g = Gpre + h@Whh^T via WMMA (units = 2 Mtiles x H/16 hidden tiles;
// each unit owns all 4 gate tiles so i/f/g/o align per lane).
// ---------------------------------------------------------------------------
template<int H>
__global__ void __launch_bounds__(256)
lstm_seq(const float* __restrict__ Gpre, const __bf16* __restrict__ Whh,
         float* __restrict__ Hout)
{
  constexpr int S = 256;
  constexpr int G4 = 4 * H;
  constexpr int UNITS = 2 * (H / 16);
  constexpr int UPW = (UNITS + 7) / 8;
  __shared__ __bf16 hbf[32 * H];
  __shared__ float  cst[32 * H];
  const int lane = threadIdx.x & 31;
  const int wave = __builtin_amdgcn_readfirstlane((int)(threadIdx.x >> 5)); // SGPR
  for (int i = threadIdx.x; i < 32 * H; i += 256) { hbf[i] = f2bf(0.0f); cst[i] = 0.0f; }
  __syncthreads();
  const int khalf = (lane >> 4) << 4;

  for (int s = 0; s < S; ++s) {
    v8f acc[UPW][4];
#pragma unroll
    for (int ui = 0; ui < UPW; ++ui) {
      const int u  = wave + ui * 8;
      const int mt = u & 1, jt = u >> 1;
      const int colj = (jt << 4) + (lane & 15);
      const int mrow = (mt << 4) + ((lane >> 4) << 3);
#pragma unroll
      for (int g = 0; g < 4; ++g)
#pragma unroll
        for (int r = 0; r < 8; ++r)
          acc[ui][g][r] = Gpre[(size_t)((mrow + r) * 256 + s) * G4 + g * H + colj];

      const int arow = (mt << 4) + (lane & 15);
#pragma unroll
      for (int kt = 0; kt < H / 32; ++kt) {
        const int k0 = kt << 5;
        v8bf lo = *(const v8bf*)&hbf[arow * H + k0 + khalf];        // ds_load_b128
        v8bf hi = *(const v8bf*)&hbf[arow * H + k0 + 8 + khalf];    // ds_load_b128
        v16bf a = __builtin_shufflevector(lo, hi, 0, 1, 2, 3, 4, 5, 6, 7,
                                          8, 9, 10, 11, 12, 13, 14, 15);
#pragma unroll
        for (int g = 0; g < 4; ++g) {
          const __bf16* bp = Whh + (size_t)(g * H + colj) * H + k0 + ((lane >> 4) << 4);
          v16bf b = *(const v16bf*)bp;
          acc[ui][g] = __builtin_amdgcn_wmma_f32_16x16x32_bf16(
              false, a, false, b, (short)0, acc[ui][g], false, false);
        }
      }
    }
    __syncthreads();   // all reads of hbf done before updates
#pragma unroll
    for (int ui = 0; ui < UPW; ++ui) {
      const int u  = wave + ui * 8;
      const int mt = u & 1, jt = u >> 1;
      const int colj = (jt << 4) + (lane & 15);
      const int mrow = (mt << 4) + ((lane >> 4) << 3);
#pragma unroll
      for (int r = 0; r < 8; ++r) {
        const int m = mrow + r;
        float iv = sigmf(acc[ui][0][r]);
        float fv = sigmf(acc[ui][1][r]);
        float gv = tanhf(acc[ui][2][r]);
        float ov = sigmf(acc[ui][3][r]);
        float cn = fv * cst[m * H + colj] + iv * gv;
        cst[m * H + colj] = cn;
        float hv = ov * tanhf(cn);
        hbf[m * H + colj] = f2bf(hv);
        Hout[(size_t)(m * 256 + s) * H + colj] = hv;
      }
    }
    __syncthreads();   // updates visible before next step's reads
  }
}

__global__ void concat_last(const float* __restrict__ h1, const float* __restrict__ h2,
                            float* __restrict__ z) {
  int idx = blockIdx.x * blockDim.x + threadIdx.x;
  if (idx >= 32 * 192) return;
  int m = idx / 192, j = idx % 192;
  z[idx] = (j < 128) ? h1[(size_t)(m * 256 + 255) * 128 + j]
                     : h2[(size_t)(m * 256 + 255) * 64 + (j - 128)];
}
__global__ void logsoftmax20(const float* __restrict__ z, float* __restrict__ out) {
  int m = threadIdx.x;
  if (m >= 32) return;
  float mx = -3.4e38f;
  for (int j = 0; j < 20; ++j) mx = fmaxf(mx, z[m * 20 + j]);
  float ss = 0.0f;
  for (int j = 0; j < 20; ++j) ss += __expf(z[m * 20 + j] - mx);
  float lg = __logf(ss);
  for (int j = 0; j < 20; ++j) out[m * 20 + j] = z[m * 20 + j] - mx - lg;
}

// ---------------------------------------------------------------------------
extern "C" void kernel_launch(void* const* d_in, const int* in_sizes, int n_in,
                              void* d_out, int out_size, void* d_ws, size_t ws_size,
                              hipStream_t stream)
{
  (void)in_sizes; (void)n_in; (void)out_size; (void)ws_size;
  constexpr int T = 32, Bg = 256, Nj = 25, NNc = Bg * Nj, E = Bg * 48, NE = E + NNc;

  const float* x  = (const float*)d_in[0];
  const int*   ei = (const int*)d_in[1];
  const float* W1 = (const float*)d_in[3]; const float* b1 = (const float*)d_in[4];
  const float* W2 = (const float*)d_in[5]; const float* b2 = (const float*)d_in[6];
  const float* W3 = (const float*)d_in[7]; const float* b3 = (const float*)d_in[8];
  const float* L1[3][4]; const float* L2[3][4];
  for (int l = 0; l < 3; ++l)
    for (int p = 0; p < 4; ++p) {
      L1[l][p] = (const float*)d_in[9  + 4 * l + p];
      L2[l][p] = (const float*)d_in[21 + 4 * l + p];
    }
  const float* fc1W = (const float*)d_in[33]; const float* fc1b = (const float*)d_in[34];
  const float* fc2W = (const float*)d_in[35]; const float* fc2b = (const float*)d_in[36];
  const float* fc3W = (const float*)d_in[37]; const float* fc3b = (const float*)d_in[38];

  uintptr_t cur = (uintptr_t)d_ws;
  auto alloc = [&](size_t bytes) -> void* {
    cur = (cur + 255) & ~(uintptr_t)255;
    void* p = (void*)cur; cur += bytes; return p;
  };
  float* deg    = (float*)alloc((size_t)NNc * 4);
  float* nrm    = (float*)alloc((size_t)NE * 4);
  float* Xp     = (float*)alloc((size_t)T * NNc * 32 * 4);   // padded features
  float* Mb     = (float*)alloc((size_t)NNc * 256 * 4);
  float* A1     = (float*)alloc((size_t)NNc * 64 * 4);
  float* A2     = (float*)alloc((size_t)NNc * 128 * 4);
  float* A3     = (float*)alloc((size_t)NNc * 256 * 4);
  float* pooled = (float*)alloc((size_t)T * Bg * 256 * 4);
  float* Gpre   = (float*)alloc((size_t)T * Bg * 512 * 4);
  float* H1a    = (float*)alloc((size_t)T * Bg * 128 * 4);
  float* H1b    = (float*)alloc((size_t)T * Bg * 128 * 4);
  float* H2a    = (float*)alloc((size_t)T * Bg * 64 * 4);
  float* H2b    = (float*)alloc((size_t)T * Bg * 64 * 4);
  float* Z0 = (float*)alloc(32 * 192 * 4);
  float* Z1 = (float*)alloc(32 * 128 * 4);
  float* Z2 = (float*)alloc(32 * 64 * 4);
  float* Z3 = (float*)alloc(32 * 20 * 4);
  __bf16* Wt1 = (__bf16*)alloc((size_t)64 * 32 * 2);
  __bf16* Wt2 = (__bf16*)alloc((size_t)128 * 64 * 2);
  __bf16* Wt3 = (__bf16*)alloc((size_t)256 * 128 * 2);
  const int ins1[3] = {256, 128, 128};
  const int ins2[3] = {256, 64, 64};
  __bf16 *Wih1[3], *Whh1[3], *Wih2[3], *Whh2[3]; float *gb1[3], *gb2[3];
  for (int l = 0; l < 3; ++l) {
    Wih1[l] = (__bf16*)alloc((size_t)512 * ins1[l] * 2);
    Whh1[l] = (__bf16*)alloc((size_t)512 * 128 * 2);
    gb1[l]  = (float*)alloc(512 * 4);
    Wih2[l] = (__bf16*)alloc((size_t)256 * ins2[l] * 2);
    Whh2[l] = (__bf16*)alloc((size_t)256 * 64 * 2);
    gb2[l]  = (float*)alloc(256 * 4);
  }
  __bf16* fc1t = (__bf16*)alloc((size_t)128 * 192 * 2);
  __bf16* fc2t = (__bf16*)alloc((size_t)64 * 128 * 2);
  __bf16* fc3t = (__bf16*)alloc((size_t)32 * 64 * 2);

  auto prep = [&](const float* W, __bf16* Bt, int K, int N, int Kp, int Np, int direct) {
    int n = Kp * Np;
    prep_weight<<<(n + 255) / 256, 256, 0, stream>>>(W, Bt, K, N, Kp, Np, direct);
  };
  // A must have rows of exactly K floats, K % 32 == 0, M % 16 == 0.
  auto gemm4 = [&](const float* A, const __bf16* Bt, const float* bias, float* C,
                   int M, int K, int N, int Np, int relu) {
    int units = (M / 16) * (Np / 64);
    wmma_gemm<4><<<(units + 3) / 4, 128, 0, stream>>>(A, Bt, bias, C, M, K, N, Np, relu);
  };
  auto gemm2 = [&](const float* A, const __bf16* Bt, const float* bias, float* C,
                   int M, int K, int N, int Np, int relu) {
    int units = (M / 16) * (Np / 32);
    wmma_gemm<2><<<(units + 3) / 4, 128, 0, stream>>>(A, Bt, bias, C, M, K, N, Np, relu);
  };

  // --- weight + input prep ---
  prep(W1, Wt1, 5, 64, 32, 64, 0);
  prep(W2, Wt2, 64, 128, 64, 128, 0);
  prep(W3, Wt3, 128, 256, 128, 256, 0);
  for (int l = 0; l < 3; ++l) {
    prep(L1[l][0], Wih1[l], ins1[l], 512, ins1[l], 512, 1);
    prep(L1[l][1], Whh1[l], 128, 512, 128, 512, 1);
    add2<<<2, 256, 0, stream>>>(L1[l][2], L1[l][3], gb1[l], 512);
    prep(L2[l][0], Wih2[l], ins2[l], 256, ins2[l], 256, 1);
    prep(L2[l][1], Whh2[l], 64, 256, 64, 256, 1);
    add2<<<1, 256, 0, stream>>>(L2[l][2], L2[l][3], gb2[l], 256);
  }
  prep(fc1W, fc1t, 192, 128, 192, 128, 0);
  prep(fc2W, fc2t, 128, 64, 128, 64, 0);
  prep(fc3W, fc3t, 64, 20, 64, 32, 0);
  pad_x5<<<((T * NNc * 32) + 255) / 256, 256, 0, stream>>>(x, Xp, T * NNc);

  // --- graph normalization (shared across frames) ---
  zero_f<<<(NNc + 255) / 256, 256, 0, stream>>>(deg, NNc);
  count_deg<<<(NE + 255) / 256, 256, 0, stream>>>(deg, ei + E, E, NNc);
  edge_norm<<<(NE + 255) / 256, 256, 0, stream>>>(nrm, deg, ei, ei + E, E, NNc);

  // --- GCN per frame ---
  for (int t = 0; t < T; ++t) {
    const float* xt = Xp + (size_t)t * NNc * 32;
    gemm4(xt, Wt1, nullptr, Mb, NNc, 32, 64, 64, 0);
    fill_bias<<<(NNc * 64 + 255) / 256, 256, 0, stream>>>(A1, b1, NNc, 64);
    scatter_edges<<<(NE * 64 + 255) / 256, 256, 0, stream>>>(Mb, A1, ei, ei + E, nrm, NE, E, 64);
    relu_ip<<<(NNc * 64 + 255) / 256, 256, 0, stream>>>(A1, NNc * 64);

    gemm4(A1, Wt2, nullptr, Mb, NNc, 64, 128, 128, 0);
    fill_bias<<<(NNc * 128 + 255) / 256, 256, 0, stream>>>(A2, b2, NNc, 128);
    scatter_edges<<<(NE * 128 + 255) / 256, 256, 0, stream>>>(Mb, A2, ei, ei + E, nrm, NE, E, 128);
    relu_ip<<<(NNc * 128 + 255) / 256, 256, 0, stream>>>(A2, NNc * 128);

    gemm4(A2, Wt3, nullptr, Mb, NNc, 128, 256, 256, 0);
    fill_bias<<<(NNc * 256 + 255) / 256, 256, 0, stream>>>(A3, b3, NNc, 256);
    scatter_edges<<<(NE * 256 + 255) / 256, 256, 0, stream>>>(Mb, A3, ei, ei + E, nrm, NE, E, 256);
    relu_ip<<<(NNc * 256 + 255) / 256, 256, 0, stream>>>(A3, NNc * 256);

    pool_frame<<<(Bg * 256 + 255) / 256, 256, 0, stream>>>(
        A3, pooled + (size_t)t * Bg * 256, Bg, Nj, 256);
  }

  // --- LSTM stack 1 (H=128) ---
  const float* in1 = pooled; int k1 = 256;
  float* hp1[2] = {H1a, H1b};
  for (int l = 0; l < 3; ++l) {
    gemm4(in1, Wih1[l], gb1[l], Gpre, T * Bg, k1, 512, 512, 0);
    lstm_seq<128><<<1, 256, 0, stream>>>(Gpre, Whh1[l], hp1[l & 1]);
    in1 = hp1[l & 1]; k1 = 128;
  }
  const float* h1fin = in1;

  // --- LSTM stack 2 (H=64) ---
  const float* in2 = pooled; int k2 = 256;
  float* hp2[2] = {H2a, H2b};
  for (int l = 0; l < 3; ++l) {
    gemm4(in2, Wih2[l], gb2[l], Gpre, T * Bg, k2, 256, 256, 0);
    lstm_seq<64><<<1, 256, 0, stream>>>(Gpre, Whh2[l], hp2[l & 1]);
    in2 = hp2[l & 1]; k2 = 64;
  }
  const float* h2fin = in2;

  // --- head ---
  concat_last<<<(32 * 192 + 255) / 256, 256, 0, stream>>>(h1fin, h2fin, Z0);
  gemm4(Z0, fc1t, fc1b, Z1, 32, 192, 128, 128, 1);
  gemm4(Z1, fc2t, fc2b, Z2, 32, 128, 64, 64, 1);
  gemm2(Z2, fc3t, fc3b, Z3, 32, 64, 20, 32, 0);
  logsoftmax20<<<1, 32, 0, stream>>>(Z3, (float*)d_out);
}
